// FraudGNN_30717606101624
// MI455X (gfx1250) — compile-verified, matrix operand order checked by reading, not run on hardware
//
#include <hip/hip_runtime.h>
#include <hip/hip_bf16.h>

// ---------------------------------------------------------------------------
// FraudGNN (3x GAT + BN/ReLU/residual + MLP head) for MI455X (gfx1250).
// GEMMs run on v_wmma_f32_16x16x32_bf16 (wave32, 16x16 tiles, f32 accum).
// Columns are split across waves (8 tiles = 64 acc VGPRs per wave) so the
// accumulators stay register-resident (no scratch spills).
// Edge softmax/aggregation: one wave per edge, f32 global atomics.
// ---------------------------------------------------------------------------

#define NODES_WIDTH 256
#define HEADS 4
#define HID 64
#define IN_F 128
#define NEG_SLOPE 0.2f
#define BN_EPS 1e-5f

typedef __attribute__((ext_vector_type(16))) __bf16 v16bf;
typedef __attribute__((ext_vector_type(8)))  __bf16 v8bf;
typedef __attribute__((ext_vector_type(8)))  float  v8f;

// ---- ordered-uint encoding so atomicMax on u32 == float max (handles neg) ----
__device__ __forceinline__ unsigned f2ord(float f) {
    unsigned u = __float_as_uint(f);
    return u ^ ((unsigned)((int)u >> 31) | 0x80000000u);
}
__device__ __forceinline__ float ord2f(unsigned e) {
    unsigned u = e ^ ((unsigned)((~(int)e) >> 31) | 0x80000000u);
    return __uint_as_float(u);
}
__device__ __forceinline__ float leaky(float v) { return v > 0.f ? v : NEG_SLOPE * v; }

// ---------------------------------------------------------------------------
// Elementwise f32 -> bf16 cast
// ---------------------------------------------------------------------------
__global__ void cast_bf16_kernel(const float* __restrict__ in, __bf16* __restrict__ out, long n) {
    long i = (long)blockIdx.x * blockDim.x + threadIdx.x;
    long stride = (long)gridDim.x * blockDim.x;
    for (; i < n; i += stride) out[i] = (__bf16)in[i];
}

// W[K,M] f32 (row-major) -> Wt[M,K] bf16 (row-major), i.e. weight transpose+cast
__global__ void transpose_cast_kernel(const float* __restrict__ W, __bf16* __restrict__ Wt,
                                      int K, int M) {
    long i = (long)blockIdx.x * blockDim.x + threadIdx.x;
    if (i >= (long)K * M) return;
    int k = (int)(i / M), m = (int)(i % M);
    Wt[(size_t)m * K + k] = (__bf16)W[i];
}

// ---------------------------------------------------------------------------
// WMMA GEMM: C[N,M] = A[N,K](bf16) x Wt[M,K]^T(bf16), optional bias+relu.
// 128-thread block = 4 waves = 2 row-tiles x COLSPLIT column groups.
// Each wave owns NCT_TOTAL/COLSPLIT 16x16 tiles (64 acc VGPRs for M=256).
// B is staged per-32-K-slice in LDS with padding (KP2=40) -> conflict-free
// ds_load_b128 fragment reads shared by all 4 waves.
// ---------------------------------------------------------------------------
template <int NCT_TOTAL, int COLSPLIT>
__global__ __launch_bounds__(128, 1)
void wmma_gemm_kernel(const __bf16* __restrict__ A, const __bf16* __restrict__ Wt,
                      const float* __restrict__ bias, float* __restrict__ C,
                      int N, int K, int relu_flag) {
    constexpr int M = NCT_TOTAL * 16;
    constexpr int NCT = NCT_TOTAL / COLSPLIT;   // tiles per wave
    constexpr int KP2 = 40;                     // 32 K-slice + 8 pad (bank rotate)
    __shared__ __bf16 smem[M * KP2];

    const int tid  = threadIdx.x;
    const int lane = tid & 31;
    const int wave = tid >> 5;
    const int wavesPerBlock = blockDim.x >> 5;
    const int rowTilesPerBlock = wavesPerBlock / COLSPLIT;
    const int rowTile = blockIdx.x * rowTilesPerBlock + (wave / COLSPLIT);
    const int colTileBase = (wave % COLSPLIT) * NCT;
    const int numRowTiles = (N + 15) >> 4;
    const bool active = rowTile < numRowTiles;

    int rA = rowTile * 16 + (lane & 15);
    if (rA >= N) rA = N - 1;                // clamp; stores are guarded
    const int kb = (lane >> 4) * 8;         // lane-half K base (ISA A/B layout)

    v8f acc[NCT];
#pragma unroll
    for (int t = 0; t < NCT; ++t) acc[t] = v8f{0.f, 0.f, 0.f, 0.f, 0.f, 0.f, 0.f, 0.f};

    for (int ks = 0; ks < K; ks += 32) {
        // ---- stage B slice: Wt[:, ks..ks+32) -> smem[col*KP2 + kk] ----
        __syncthreads();
        for (int idx = tid; idx < M * 4; idx += blockDim.x) {   // 16B chunks
            int col = idx >> 2, ch = idx & 3;
            *(v8bf*)(smem + col * KP2 + ch * 8) =
                *(const v8bf*)(Wt + (size_t)col * K + ks + ch * 8);
        }
        __syncthreads();

        // ---- A fragment: row rA, K = ks+kb..+7 and ks+kb+16..+23 ----
        const __bf16* ap = A + (size_t)rA * K + ks + kb;
        v8bf alo = *(const v8bf*)(ap);
        v8bf ahi = *(const v8bf*)(ap + 16);
        __builtin_prefetch(ap + 32, 0, 0);
        v16bf afrag;
#pragma unroll
        for (int i = 0; i < 8; ++i) { afrag[i] = alo[i]; afrag[i + 8] = ahi[i]; }

#pragma unroll
        for (int t = 0; t < NCT; ++t) {
            const __bf16* bp = smem + ((colTileBase + t) * 16 + (lane & 15)) * KP2 + kb;
            v8bf blo = *(const v8bf*)(bp);
            v8bf bhi = *(const v8bf*)(bp + 16);
            v16bf bfrag;
#pragma unroll
            for (int i = 0; i < 8; ++i) { bfrag[i] = blo[i]; bfrag[i + 8] = bhi[i]; }
            acc[t] = __builtin_amdgcn_wmma_f32_16x16x32_bf16(
                false, afrag, false, bfrag, (short)0, acc[t], false, false);
        }
    }

    if (!active) return;
    const int colLane = lane & 15;
    const int rbase = rowTile * 16 + (lane >> 4) * 8;
#pragma unroll
    for (int t = 0; t < NCT; ++t) {
        int col = (colTileBase + t) * 16 + colLane;
        float bval = bias ? bias[col] : 0.f;
#pragma unroll
        for (int r = 0; r < 8; ++r) {
            int row = rbase + r;
            if (row < N) {
                float v = acc[t][r] + bval;
                if (relu_flag) v = fmaxf(v, 0.f);
                C[(size_t)row * M + col] = v;
            }
        }
    }
}

// ---------------------------------------------------------------------------
// alpha_s[n,h] = <h[n,h,:], a_src[h,:]>, alpha_d likewise
// ---------------------------------------------------------------------------
__global__ void alphas_kernel(const float* __restrict__ h, const float* __restrict__ asrc,
                              const float* __restrict__ adst, float* __restrict__ out_s,
                              float* __restrict__ out_d, int N) {
    int i = blockIdx.x * blockDim.x + threadIdx.x;
    if (i >= N * HEADS) return;
    int n = i >> 2, hd = i & 3;
    const float4* hp = (const float4*)(h + (size_t)n * NODES_WIDTH + hd * HID);
    const float4* sp = (const float4*)(asrc + hd * HID);
    const float4* dp = (const float4*)(adst + hd * HID);
    float s = 0.f, d = 0.f;
#pragma unroll
    for (int j = 0; j < HID / 4; ++j) {
        float4 hv = hp[j], sv = sp[j], dv = dp[j];
        s += hv.x * sv.x + hv.y * sv.y + hv.z * sv.z + hv.w * sv.w;
        d += hv.x * dv.x + hv.y * dv.y + hv.z * dv.z + hv.w * dv.w;
    }
    out_s[i] = s;
    out_d[i] = d;
}

// zero aggregation buffer, init softmax max (ordered -inf), denom, BN sums
__global__ void init_kernel(float* __restrict__ h_agg, unsigned* __restrict__ m,
                            float* __restrict__ denom, float* __restrict__ sums,
                            long total, long n4) {
    long i = (long)blockIdx.x * blockDim.x + threadIdx.x;
    long stride = (long)gridDim.x * blockDim.x;
    unsigned neg_inf_ord = f2ord(-__builtin_inff());
    for (; i < total; i += stride) {
        h_agg[i] = 0.f;
        if (i < n4) { m[i] = neg_inf_ord; denom[i] = 0.f; }
        if (i < 2 * NODES_WIDTH) sums[i] = 0.f;
    }
}

__device__ __forceinline__ void edge_decode(const long long* __restrict__ ei, long e, int E,
                                            int& src, int& dst) {
    if (e < E) { src = (int)ei[e]; dst = (int)ei[E + e]; }
    else       { src = dst = (int)(e - E); }   // self loops appended
}

__global__ void edge_max_kernel(const long long* __restrict__ ei, const float* __restrict__ as,
                                const float* __restrict__ ad, unsigned* __restrict__ m,
                                int E, int N) {
    long i = (long)blockIdx.x * blockDim.x + threadIdx.x;
    long tot = (long)(E + N) * HEADS;
    if (i >= tot) return;
    long e = i >> 2; int hd = (int)(i & 3);
    int src, dst; edge_decode(ei, e, E, src, dst);
    float v = leaky(as[src * HEADS + hd] + ad[dst * HEADS + hd]);
    atomicMax(&m[dst * HEADS + hd], f2ord(v));
}

__global__ void edge_sum_kernel(const long long* __restrict__ ei, const float* __restrict__ as,
                                const float* __restrict__ ad, const unsigned* __restrict__ m,
                                float* __restrict__ denom, int E, int N) {
    long i = (long)blockIdx.x * blockDim.x + threadIdx.x;
    long tot = (long)(E + N) * HEADS;
    if (i >= tot) return;
    long e = i >> 2; int hd = (int)(i & 3);
    int src, dst; edge_decode(ei, e, E, src, dst);
    float v = leaky(as[src * HEADS + hd] + ad[dst * HEADS + hd]);
    atomicAdd(&denom[dst * HEADS + hd], __expf(v - ord2f(m[dst * HEADS + hd])));
}

// One wave32 per edge: 32 lanes x 8 floats = 256 channels.
__global__ void edge_agg_kernel(const long long* __restrict__ ei, const float* __restrict__ as,
                                const float* __restrict__ ad, const unsigned* __restrict__ m,
                                const float* __restrict__ denom, const float* __restrict__ hfeat,
                                float* __restrict__ out, int E, int N) {
    int lane = threadIdx.x & 31;
    int wave = threadIdx.x >> 5;
    long e = (long)blockIdx.x * (blockDim.x >> 5) + wave;
    long tot = (long)E + N;
    if (e >= tot) return;
    int src, dst; edge_decode(ei, e, E, src, dst);
    int hd = lane >> 3;                      // 8 lanes per head
    float v = leaky(as[src * HEADS + hd] + ad[dst * HEADS + hd]);
    float alpha = __expf(v - ord2f(m[dst * HEADS + hd])) /
                  (denom[dst * HEADS + hd] + 1e-16f);
    int c0 = lane * 8;
    const float4* hp = (const float4*)(hfeat + (size_t)src * NODES_WIDTH + c0);
    float4 h0 = hp[0], h1 = hp[1];
    float* op = out + (size_t)dst * NODES_WIDTH + c0;
    atomicAdd(op + 0, h0.x * alpha); atomicAdd(op + 1, h0.y * alpha);
    atomicAdd(op + 2, h0.z * alpha); atomicAdd(op + 3, h0.w * alpha);
    atomicAdd(op + 4, h1.x * alpha); atomicAdd(op + 5, h1.y * alpha);
    atomicAdd(op + 6, h1.z * alpha); atomicAdd(op + 7, h1.w * alpha);
}

// per-block register accumulation of column sums / sums of squares
__global__ void bn_stats_kernel(const float* __restrict__ h, float* __restrict__ sums, int N) {
    int col = threadIdx.x;                   // blockDim.x == 256
    int r0 = blockIdx.x * 64;
    int rend = min(r0 + 64, N);
    float s = 0.f, sq = 0.f;
    for (int r = r0; r < rend; ++r) {
        float v = h[(size_t)r * NODES_WIDTH + col];
        s += v; sq += v * v;
    }
    atomicAdd(&sums[col], s);
    atomicAdd(&sums[NODES_WIDTH + col], sq);
}

// h_act = relu(bn(h_agg)) [+ residual], also emit bf16 copy for next GEMM
__global__ void bn_apply_kernel(const float* __restrict__ h_agg, const float* __restrict__ sums,
                                const float* __restrict__ gamma, const float* __restrict__ beta,
                                float* __restrict__ h_act, __bf16* __restrict__ a_bf,
                                int N, int residual) {
    long i = (long)blockIdx.x * blockDim.x + threadIdx.x;
    if (i >= (long)N * NODES_WIDTH) return;
    int c = (int)(i & (NODES_WIDTH - 1));
    float invN = 1.f / (float)N;
    float mu = sums[c] * invN;
    float var = sums[NODES_WIDTH + c] * invN - mu * mu;
    float rstd = rsqrtf(var + BN_EPS);
    float y = gamma[c] * (h_agg[i] - mu) * rstd + beta[c];
    y = fmaxf(y, 0.f);
    if (residual) y += h_act[i];
    h_act[i] = y;
    a_bf[i] = (__bf16)y;
}

// fc2 (64->32, relu) + fc3 (32->1) + sigmoid, one thread per node
__global__ void head_kernel(const float* __restrict__ h1, const float* __restrict__ fc2w,
                            const float* __restrict__ fc2b, const float* __restrict__ fc3w,
                            const float* __restrict__ fc3b, float* __restrict__ out, int N) {
    int n = blockIdx.x * blockDim.x + threadIdx.x;
    if (n >= N) return;
    const float* hp = h1 + (size_t)n * 64;
    float hv[64];
#pragma unroll
    for (int k = 0; k < 64; ++k) hv[k] = hp[k];
    float a3 = 0.f;
#pragma unroll 4
    for (int j = 0; j < 32; ++j) {
        float s = fc2b[j];
#pragma unroll
        for (int k = 0; k < 64; ++k) s += hv[k] * fc2w[k * 32 + j];
        s = fmaxf(s, 0.f);
        a3 += s * fc3w[j];
    }
    a3 += fc3b[0];
    out[n] = 1.f / (1.f + __expf(-a3));
}

// ---------------------------------------------------------------------------
extern "C" void kernel_launch(void* const* d_in, const int* in_sizes, int n_in,
                              void* d_out, int out_size, void* d_ws, size_t ws_size,
                              hipStream_t stream) {
    (void)n_in; (void)out_size; (void)ws_size;
    const int N = in_sizes[0] / IN_F;        // 50000
    const int E = in_sizes[1] / 2;           // 800000

    const float*     x     = (const float*)d_in[0];
    const long long* ei    = (const long long*)d_in[1];
    const float*     Ws[3]    = {(const float*)d_in[2],  (const float*)d_in[3],  (const float*)d_in[4]};
    const float*     asrc[3]  = {(const float*)d_in[5],  (const float*)d_in[6],  (const float*)d_in[7]};
    const float*     adst[3]  = {(const float*)d_in[8],  (const float*)d_in[9],  (const float*)d_in[10]};
    const float*     gam[3]   = {(const float*)d_in[14], (const float*)d_in[15], (const float*)d_in[16]};
    const float*     bet[3]   = {(const float*)d_in[17], (const float*)d_in[18], (const float*)d_in[19]};
    const float* fc1_w = (const float*)d_in[20];
    const float* fc1_b = (const float*)d_in[21];
    const float* fc2_w = (const float*)d_in[22];
    const float* fc2_b = (const float*)d_in[23];
    const float* fc3_w = (const float*)d_in[24];
    const float* fc3_b = (const float*)d_in[25];

    // ---- workspace carve-up (256B-aligned) ----
    char* ws = (char*)d_ws;
    size_t off = 0;
    auto carve = [&](size_t bytes) { char* p = ws + off; off += (bytes + 255) & ~(size_t)255; return p; };
    __bf16*   a_bf    = (__bf16*)  carve((size_t)N * NODES_WIDTH * sizeof(__bf16));
    __bf16*   wt_bf   = (__bf16*)  carve((size_t)NODES_WIDTH * NODES_WIDTH * sizeof(__bf16));
    float*    h_lin   = (float*)   carve((size_t)N * NODES_WIDTH * sizeof(float));
    float*    h_agg   = (float*)   carve((size_t)N * NODES_WIDTH * sizeof(float));
    float*    h_act   = (float*)   carve((size_t)N * NODES_WIDTH * sizeof(float));
    float*    alpha_s = (float*)   carve((size_t)N * HEADS * sizeof(float));
    float*    alpha_d = (float*)   carve((size_t)N * HEADS * sizeof(float));
    unsigned* mmax    = (unsigned*)carve((size_t)N * HEADS * sizeof(unsigned));
    float*    denom   = (float*)   carve((size_t)N * HEADS * sizeof(float));
    float*    sums    = (float*)   carve(2 * NODES_WIDTH * sizeof(float));
    float*    h1      = (float*)   carve((size_t)N * 64 * sizeof(float));

    const long totalNW = (long)N * NODES_WIDTH;
    const long n4 = (long)N * HEADS;
    const long eh = (long)(E + N) * HEADS;
    const int numRowTiles = (N + 15) >> 4;
    const int gemmBlocks = (numRowTiles + 1) / 2;          // 2 row-tiles / block

    for (int layer = 0; layer < 3; ++layer) {
        const int K = (layer == 0) ? IN_F : NODES_WIDTH;
        // input -> bf16
        if (layer == 0)
            cast_bf16_kernel<<<2048, 256, 0, stream>>>(x, a_bf, (long)N * IN_F);
        // weights -> bf16 transposed [256, K]
        transpose_cast_kernel<<<((K * NODES_WIDTH) + 255) / 256, 256, 0, stream>>>(
            Ws[layer], wt_bf, K, NODES_WIDTH);
        // h_lin = A @ W   (v_wmma_f32_16x16x32_bf16)
        wmma_gemm_kernel<16, 2><<<gemmBlocks, 128, 0, stream>>>(
            a_bf, wt_bf, nullptr, h_lin, N, K, 0);
        // attention logits
        alphas_kernel<<<(N * HEADS + 255) / 256, 256, 0, stream>>>(
            h_lin, asrc[layer], adst[layer], alpha_s, alpha_d, N);
        // clear aggregation / softmax state / BN sums
        init_kernel<<<(int)((totalNW + 255) / 256), 256, 0, stream>>>(
            h_agg, mmax, denom, sums, totalNW, n4);
        // per-dst softmax over edges (+self loops)
        edge_max_kernel<<<(int)((eh + 255) / 256), 256, 0, stream>>>(ei, alpha_s, alpha_d, mmax, E, N);
        edge_sum_kernel<<<(int)((eh + 255) / 256), 256, 0, stream>>>(ei, alpha_s, alpha_d, mmax, denom, E, N);
        // message aggregation: one wave per edge
        edge_agg_kernel<<<(int)(((long)E + N + 7) / 8), 256, 0, stream>>>(
            ei, alpha_s, alpha_d, mmax, denom, h_lin, h_agg, E, N);
        // batch norm + relu (+ residual for layers 1,2); also refresh a_bf
        bn_stats_kernel<<<(N + 63) / 64, 256, 0, stream>>>(h_agg, sums, N);
        bn_apply_kernel<<<(int)((totalNW + 255) / 256), 256, 0, stream>>>(
            h_agg, sums, gam[layer], bet[layer], h_act, a_bf, N, layer > 0 ? 1 : 0);
    }

    // fc1: [N,256] @ [256,64] + bias, relu  (WMMA, 4 col-tiles, 2 per wave)
    transpose_cast_kernel<<<((NODES_WIDTH * 64) + 255) / 256, 256, 0, stream>>>(
        fc1_w, wt_bf, NODES_WIDTH, 64);
    wmma_gemm_kernel<4, 2><<<gemmBlocks, 128, 0, stream>>>(
        a_bf, wt_bf, fc1_b, h1, N, NODES_WIDTH, 1);

    // fc2 + fc3 + sigmoid
    head_kernel<<<(N + 255) / 256, 256, 0, stream>>>(
        h1, fc2_w, fc2_b, fc3_w, fc3_b, (float*)d_out, N);
}
